// ConvMPN_53214644797504
// MI455X (gfx1250) — matrix operand order repaired and checked by reading, not compiled, not run on hardware
//
#include <hip/hip_runtime.h>

typedef __attribute__((ext_vector_type(16))) _Float16 v16h;
typedef __attribute__((ext_vector_type(8)))  float    v8f;

#define HW 1024
#define IMG 32

__device__ __forceinline__ v8f wmma16x16x32(v16h a, v16h b, v8f c) {
  // D = A(16x32 f16) * B(32x16 f16) + C(16x16 f32)
  return __builtin_amdgcn_wmma_f32_16x16x32_f16(false, a, false, b, (short)0, c,
                                                false, false);
}

// ---------------------------------------------------------------- zero scratch
__global__ void zero_ws_kernel(float4* __restrict__ p, long n4) {
  long i = (long)blockIdx.x * blockDim.x + threadIdx.x;
  long stride = (long)gridDim.x * blockDim.x;
  float4 z = make_float4(0.f, 0.f, 0.f, 0.f);
  for (; i < n4; i += stride) p[i] = z;
}

// ------------------------------------------------------------- edge scatter
// neigh[b] += x[a]; neigh[a] += x[b]  (rel>0), same for non_neigh (rel<0)
__global__ void edge_scatter_kernel(const float* __restrict__ x,
                                    const long long* __restrict__ edges,
                                    float* __restrict__ neigh,
                                    float* __restrict__ nonn) {
  const int e = blockIdx.x;
  const long long a = edges[3 * e + 0];
  const long long r = edges[3 * e + 1];
  const long long b = edges[3 * e + 2];
  float* dst = (r > 0) ? neigh : nonn;
  const float* xa = x + a * (16 * HW);
  const float* xb = x + b * (16 * HW);
  float* da = dst + a * (16 * HW);
  float* db = dst + b * (16 * HW);
  for (int i = threadIdx.x; i < 16 * HW; i += blockDim.x) {
    atomicAdd(db + i, xa[i]);
    atomicAdd(da + i, xb[i]);
  }
}

// ----------------------------------------------- build WMMA A-fragments (f16)
// 16-bit A 16x32 layout (contiguous k, matching B fills):
//   lane<16 : M=lane,    K=2v+h
//   lane>=16: M=lane-16, K=16+2v+h
// frag ids: conv1: mtile*18+shift*2+kstep (0..35, Cin=48 padded to 64)
//           conv2: 36 + mtile*9 + shift   (Cin=32)
//           conv3: 54 + shift             (Cin=32, Cout=16)
__global__ void prep_afrags_kernel(const float* __restrict__ w1,
                                   const float* __restrict__ w2,
                                   const float* __restrict__ w3,
                                   _Float16* __restrict__ frags) {
  int t = blockIdx.x * blockDim.x + threadIdx.x;
  if (t >= 63 * 32) return;
  const int frag = t >> 5;
  const int lane = t & 31;
  const float* w;
  int Cin, mtile, shift, kstep;
  if (frag < 36) {
    w = w1; Cin = 48;
    mtile = frag / 18;
    int r = frag % 18;
    shift = r >> 1;
    kstep = r & 1;
  } else if (frag < 54) {
    w = w2; Cin = 32;
    int f = frag - 36;
    mtile = f / 9; shift = f % 9; kstep = 0;
  } else {
    w = w3; Cin = 32;
    mtile = 0; shift = frag - 54; kstep = 0;
  }
  const int kh = shift / 3, kw = shift % 3;
  const int co = mtile * 16 + (lane & 15);
  _Float16* out = frags + ((long)frag * 32 + lane) * 16;
  for (int v = 0; v < 8; ++v)
    for (int h = 0; h < 2; ++h) {
      int K = 2 * v + h + ((lane < 16) ? 0 : 16);
      int ci = kstep * 32 + K;
      float val = (ci < Cin) ? w[((co * Cin + ci) * 3 + kh) * 3 + kw] : 0.0f;
      out[v * 2 + h] = (_Float16)val;
    }
}

// ------------------------------------------------------------------- conv1
// enc = [x | neigh | non_neigh] (48ch, pad->64), out 32ch f16 [n][co][pos]
// 1024 threads = 32 waves; wave owns mtile=wave&1, 4 position tiles.
__global__ __launch_bounds__(1024) void conv1_kernel(
    const float* __restrict__ x, const float* __restrict__ neigh,
    const float* __restrict__ nonn, const v16h* __restrict__ afrags,
    const float* __restrict__ bias, _Float16* __restrict__ h1) {
  extern __shared__ _Float16 lds[];  // [34][34][64] f16, halo+chan padded
  const int tid = threadIdx.x;
  const int n = blockIdx.x;
  {
    uint4 z = make_uint4(0u, 0u, 0u, 0u);
    uint4* l4 = (uint4*)lds;
    for (int i = tid; i < (34 * 34 * 64 * 2) / 16; i += 1024) l4[i] = z;
  }
  __syncthreads();
  const long nb = (long)n * (16 * HW);
  for (int idx = tid; idx < 48 * HW; idx += 1024) {
    int c = idx >> 10, pos = idx & 1023;
    int y = pos >> 5, xc = pos & 31;
    float v;
    if (c < 16)      v = x[nb + (long)c * HW + pos];
    else if (c < 32) v = neigh[nb + (long)(c - 16) * HW + pos];
    else             v = nonn[nb + (long)(c - 32) * HW + pos];
    lds[((y + 1) * 34 + (xc + 1)) * 64 + c] = (_Float16)v;
  }
  __syncthreads();
  const int lane = tid & 31, wave = tid >> 5;
  const int mtile = wave & 1;
  const int ntbase = (wave >> 1) * 4;
  const int n16 = lane & 15;
  const int hi = lane >> 4;  // B lanes 0-15: K 0..15, lanes 16-31: K 16..31

  auto boff = [&](int nt, int kh, int kw, int ks) {
    const int y = nt >> 1;
    const int xc = ((nt & 1) << 4) + n16;
    return ((y + kh) * 34 + (xc + kw)) * 64 + ks * 32 + hi * 16;
  };

  v8f acc[4];
#pragma unroll
  for (int i = 0; i < 4; ++i) acc[i] = {};

  // double-buffered A (global) and B (LDS) to overlap loads with WMMA
  v16h Acur = afrags[(mtile * 18 + 0) * 32 + lane];
  for (int sk = 0; sk < 18; ++sk) {
    v16h Anext = Acur;
    if (sk + 1 < 18) Anext = afrags[(mtile * 18 + sk + 1) * 32 + lane];
    const int shift = sk >> 1, ks = sk & 1;
    const int kh = shift / 3, kw = shift % 3;
    v16h Bcur = *(const v16h*)(lds + boff(ntbase, kh, kw, ks));
#pragma unroll
    for (int i = 0; i < 4; ++i) {
      v16h Bnext = Bcur;
      if (i + 1 < 4)
        Bnext = *(const v16h*)(lds + boff(ntbase + i + 1, kh, kw, ks));
      acc[i] = wmma16x16x32(Acur, Bcur, acc[i]);
      Bcur = Bnext;
    }
    Acur = Anext;
  }

#pragma unroll
  for (int i = 0; i < 4; ++i) {
    const int nt = ntbase + i;
    const int y = nt >> 1;
    const int xc = ((nt & 1) << 4) + n16;
#pragma unroll
    for (int r = 0; r < 8; ++r) {
      const int co = mtile * 16 + r + hi * 8;
      h1[((long)n * 32 + co) * HW + y * IMG + xc] =
          (_Float16)(acc[i][r] + bias[co]);
    }
  }
}

// -------------------------------------------------------------- conv2/conv3
template <int COUT, bool F32OUT>
__global__ __launch_bounds__(1024) void conv23_kernel(
    const _Float16* __restrict__ src,  // [n][32][1024] f16
    const v16h* __restrict__ afrags, int fragBase,
    const float* __restrict__ bias, void* __restrict__ dstv) {
  extern __shared__ _Float16 lds[];  // [34][34][32] f16
  const int tid = threadIdx.x;
  const int n = blockIdx.x;
  {
    uint4 z = make_uint4(0u, 0u, 0u, 0u);
    uint4* l4 = (uint4*)lds;
    for (int i = tid; i < (34 * 34 * 32 * 2) / 16; i += 1024) l4[i] = z;
  }
  __syncthreads();
  const long nb = (long)n * (32 * HW);
  for (int idx = tid; idx < 32 * HW; idx += 1024) {
    int c = idx >> 10, pos = idx & 1023;
    int y = pos >> 5, xc = pos & 31;
    lds[((y + 1) * 34 + (xc + 1)) * 32 + c] = src[nb + idx];
  }
  __syncthreads();
  const int lane = tid & 31, wave = tid >> 5;
  constexpr int MTILES = COUT / 16;
  constexpr int TPW = (MTILES * 64) / 32;  // tiles per wave: 4 or 2
  const int mtile = (MTILES == 2) ? (wave & 1) : 0;
  const int ntbase = (MTILES == 2) ? ((wave >> 1) * TPW) : (wave * TPW);
  const int n16 = lane & 15;
  const int hi = lane >> 4;

  auto boff = [&](int nt, int kh, int kw) {
    const int y = nt >> 1;
    const int xc = ((nt & 1) << 4) + n16;
    return ((y + kh) * 34 + (xc + kw)) * 32 + hi * 16;
  };

  v8f acc[TPW];
#pragma unroll
  for (int i = 0; i < TPW; ++i) acc[i] = {};

  v16h Acur = afrags[(fragBase + mtile * 9 + 0) * 32 + lane];
  for (int shift = 0; shift < 9; ++shift) {
    v16h Anext = Acur;
    if (shift + 1 < 9)
      Anext = afrags[(fragBase + mtile * 9 + shift + 1) * 32 + lane];
    const int kh = shift / 3, kw = shift % 3;
    v16h Bcur = *(const v16h*)(lds + boff(ntbase, kh, kw));
#pragma unroll
    for (int i = 0; i < TPW; ++i) {
      v16h Bnext = Bcur;
      if (i + 1 < TPW)
        Bnext = *(const v16h*)(lds + boff(ntbase + i + 1, kh, kw));
      acc[i] = wmma16x16x32(Acur, Bcur, acc[i]);
      Bcur = Bnext;
    }
    Acur = Anext;
  }

#pragma unroll
  for (int i = 0; i < TPW; ++i) {
    const int nt = ntbase + i;
    const int y = nt >> 1;
    const int xc = ((nt & 1) << 4) + n16;
#pragma unroll
    for (int r = 0; r < 8; ++r) {
      const int co = mtile * 16 + r + hi * 8;
      const float val = acc[i][r] + bias[co];
      if (F32OUT)
        ((float*)dstv)[((long)n * COUT + co) * HW + y * IMG + xc] = val;
      else
        ((_Float16*)dstv)[((long)n * COUT + co) * HW + y * IMG + xc] =
            (_Float16)val;
    }
  }
}

// ---------------------------------------------------------------------------
extern "C" void kernel_launch(void* const* d_in, const int* in_sizes, int n_in,
                              void* d_out, int out_size, void* d_ws,
                              size_t ws_size, hipStream_t stream) {
  (void)n_in; (void)out_size; (void)ws_size;
  const float*     x     = (const float*)d_in[0];
  const long long* edges = (const long long*)d_in[1];
  const float*     w1    = (const float*)d_in[2];
  const float*     b1    = (const float*)d_in[3];
  const float*     w2    = (const float*)d_in[4];
  const float*     b2    = (const float*)d_in[5];
  const float*     w3    = (const float*)d_in[6];
  const float*     b3    = (const float*)d_in[7];
  float* out = (float*)d_out;

  const int N = in_sizes[0] / (16 * HW);  // 2048
  const int E = in_sizes[1] / 3;          // 16384

  // workspace layout
  char* ws = (char*)d_ws;
  _Float16* frags = (_Float16*)ws;                       // 63*32*16*2 B
  float* neigh = (float*)(ws + (1 << 16));
  float* nonn = neigh + (long)N * 16 * HW;
  _Float16* h1 = (_Float16*)(ws + (1 << 16) + 2L * N * 16 * HW * 4);
  _Float16* h2 = (_Float16*)neigh;  // alias: neigh dead after conv1

  // 1) zero the two scatter accumulators (contiguous)
  long n4 = (2L * N * 16 * HW) / 4;
  zero_ws_kernel<<<4096, 256, 0, stream>>>((float4*)neigh, n4);

  // 2) edge scatter (f32 atomics)
  edge_scatter_kernel<<<E, 256, 0, stream>>>(x, edges, neigh, nonn);

  // 3) weight -> A-fragment reshape
  prep_afrags_kernel<<<(63 * 32 + 255) / 256, 256, 0, stream>>>(w1, w2, w3,
                                                                frags);

  // 4) conv stack (big-LDS, >64KB dynamic shared)
  const int lds1 = 34 * 34 * 64 * 2;  // 147,968 B
  const int lds2 = 34 * 34 * 32 * 2;  //  73,984 B
  hipFuncSetAttribute((const void*)conv1_kernel,
                      hipFuncAttributeMaxDynamicSharedMemorySize, lds1);
  hipFuncSetAttribute((const void*)(&conv23_kernel<32, false>),
                      hipFuncAttributeMaxDynamicSharedMemorySize, lds2);
  hipFuncSetAttribute((const void*)(&conv23_kernel<16, true>),
                      hipFuncAttributeMaxDynamicSharedMemorySize, lds2);

  conv1_kernel<<<N, 1024, lds1, stream>>>(x, neigh, nonn, (const v16h*)frags,
                                          b1, h1);
  conv23_kernel<32, false><<<N, 1024, lds2, stream>>>(
      h1, (const v16h*)frags, 36, b2, (void*)h2);
  conv23_kernel<16, true><<<N, 1024, lds2, stream>>>(
      h2, (const v16h*)frags, 54, b3, (void*)out);
}
